// BiLSTM_CRF_86663850099058
// MI455X (gfx1250) — matrix operand order repaired and checked by reading, not compiled, not run on hardware
//
#include <hip/hip_runtime.h>
#include <hip/hip_bf16.h>

typedef __attribute__((ext_vector_type(16))) _Float16 v16h;
typedef __attribute__((ext_vector_type(8)))  _Float16 v8h;
typedef __attribute__((ext_vector_type(2)))  _Float16 v2h;
typedef __attribute__((ext_vector_type(8)))  float    v8f;

#define S_LEN   4096
#define L_CH    24
#define EMB     300
#define CHAR_E  25
#define CH_OUT  25
#define HID     512
#define KDIM    325      // EMB + CH_OUT
#define KPAD    352      // padded to multiple of 32 for WMMA K
#define GATES   2048     // 4*HID, PyTorch gate order i,f,g,o
#define TAGS    11
#define NPAD    16       // tag dim padded for WMMA N
#define START_TAG 9
#define STOP_TAG  10
#define NEGV   -10000.0f
#define NSLICE  8        // WGP slices per direction (64 hidden units each)

// ---------------------------------------------------------------------------
// Weight conversion: f32 -> f16, pad Wih K to 352, transpose Whh to [k][n],
// fuse bih+bhh, pad h2t_w to [16][1024] f16.
// ---------------------------------------------------------------------------
__global__ void convert_kernel(const float* __restrict__ Wih_f, const float* __restrict__ Whh_f,
                               const float* __restrict__ bih_f, const float* __restrict__ bhh_f,
                               const float* __restrict__ Wih_b, const float* __restrict__ Whh_b,
                               const float* __restrict__ bih_b, const float* __restrict__ bhh_b,
                               const float* __restrict__ h2t_w,
                               _Float16* __restrict__ wih_h, _Float16* __restrict__ whh_t,
                               float* __restrict__ bias2, _Float16* __restrict__ h2t_h) {
  const long N1 = 2L * GATES * KPAD;
  const long N2 = 2L * HID * GATES;
  const long N3 = 2L * GATES;
  const long N4 = (long)NPAD * 2 * HID;
  const long total = N1 + N2 + N3 + N4;
  for (long i = blockIdx.x * (long)blockDim.x + threadIdx.x; i < total;
       i += (long)gridDim.x * blockDim.x) {
    if (i < N1) {
      long d = i / ((long)GATES * KPAD), rem = i % ((long)GATES * KPAD);
      int n = (int)(rem / KPAD), k = (int)(rem % KPAD);
      const float* Wsrc = d ? Wih_b : Wih_f;
      wih_h[i] = (k < KDIM) ? (_Float16)Wsrc[(long)n * KDIM + k] : (_Float16)0.0f;
    } else if (i < N1 + N2) {
      long j = i - N1;
      long d = j / ((long)HID * GATES), rem = j % ((long)HID * GATES);
      int k = (int)(rem / GATES), n = (int)(rem % GATES);
      const float* Wsrc = d ? Whh_b : Whh_f;
      whh_t[j] = (_Float16)Wsrc[(long)n * HID + k];   // [dir][k][n]
    } else if (i < N1 + N2 + N3) {
      long j = i - N1 - N2;
      long d = j / GATES; int n = (int)(j % GATES);
      bias2[j] = d ? (bih_b[n] + bhh_b[n]) : (bih_f[n] + bhh_f[n]);
    } else {
      long j = i - N1 - N2 - N3;
      int n = (int)(j / (2 * HID)), k = (int)(j % (2 * HID));
      h2t_h[j] = (n < TAGS) ? (_Float16)h2t_w[(long)n * 2 * HID + k] : (_Float16)0.0f;
    }
  }
}

// ---------------------------------------------------------------------------
// Char CNN (k=3 over length, full width, pad 2) + max-pool + word embedding
// concat -> emb_h[s][352] f16 (zero padded 325..351).
// ---------------------------------------------------------------------------
__global__ __launch_bounds__(128) void prep_kernel(const int* __restrict__ sentence,
    const int* __restrict__ chars, const float* __restrict__ word_emb,
    const float* __restrict__ char_emb, const float* __restrict__ conv_w,
    const float* __restrict__ conv_b, _Float16* __restrict__ emb_h) {
  __shared__ float ce[L_CH][CHAR_E];
  __shared__ float cf[CH_OUT];
  const int s = blockIdx.x, tid = threadIdx.x;
  for (int i = tid; i < L_CH * CHAR_E; i += 128) {
    int y = i / CHAR_E, e = i % CHAR_E;
    ce[y][e] = char_emb[chars[s * L_CH + y] * CHAR_E + e];
  }
  __syncthreads();
  if (tid < CH_OUT) {
    float m = -3.0e38f;
    for (int y = 0; y < L_CH + 2; ++y) {          // output height 26 (pad 2,2)
      float acc = conv_b[tid];
      for (int dy = 0; dy < 3; ++dy) {
        int yy = y - 2 + dy;
        if (yy >= 0 && yy < L_CH) {
          const float* wrow = conv_w + tid * 75 + dy * CHAR_E;
          #pragma unroll
          for (int e = 0; e < CHAR_E; ++e) acc += ce[yy][e] * wrow[e];
        }
      }
      m = fmaxf(m, acc);
    }
    cf[tid] = m;
  }
  __syncthreads();
  const int w = sentence[s];
  for (int i = tid; i < KPAD; i += 128) {
    float v = (i < EMB) ? word_emb[(long)w * EMB + i]
                        : ((i < KDIM) ? cf[i - EMB] : 0.0f);
    emb_h[(size_t)s * KPAD + i] = (_Float16)v;
  }
}

// ---------------------------------------------------------------------------
// WMMA fragment helpers (f16 16x16x32, this file's contiguous-K A/B order, ISA C/D layout):
// A 16x32: lane -> M=lane&15, 16 contiguous K halves, base 0/16 by half.
// B 32x16: lane -> N=lane&15, 16 contiguous K halves, base 0/16 by half.
// C/D f32: lane = N + (M>=8)*16, component = M&7 (bias per-N is uniform/lane).
// ---------------------------------------------------------------------------
__device__ __forceinline__ v16h load_a_frag(const _Float16* A, int lda, int m0, int k0, int lane) {
  int row = m0 + (lane & 15);
  int kb  = k0 + ((lane & 16) ? 16 : 0);
  v8h lo = *(const v8h*)(A + (size_t)row * lda + kb);
  v8h hi = *(const v8h*)(A + (size_t)row * lda + kb + 8);
  v16h r;
  #pragma unroll
  for (int i = 0; i < 8; ++i) { r[i] = lo[i]; r[i + 8] = hi[i]; }
  return r;
}
__device__ __forceinline__ v16h load_b_frag(const _Float16* B, int ldb, int n0, int k0, int lane) {
  int col = n0 + (lane & 15);
  int kb  = k0 + ((lane & 16) ? 16 : 0);
  return *(const v16h*)(B + (size_t)col * ldb + kb);
}
__device__ __forceinline__ void store_tile(const v8f& c, float* Gd, const float* bias,
                                           int m0, int n0, int lane) {
  int n = n0 + (lane & 15);
  float bv = bias[n];
  int mbase = m0 + ((lane & 16) ? 8 : 0);
  #pragma unroll
  for (int r = 0; r < 8; ++r) Gd[(size_t)(mbase + r) * GATES + n] = c[r] + bv;
}

// ---------------------------------------------------------------------------
// Input-projection GEMM: G[dir][t][n] = emb[t][:] . Wih[dir][n][:] + bias[n]
// 4096 x 2048 x 352, f16 operands, f32 accumulation, bias fused in epilogue.
// ---------------------------------------------------------------------------
__global__ __launch_bounds__(256) void gemm_gates_kernel(const _Float16* __restrict__ A,
    const _Float16* __restrict__ wih_h, const float* __restrict__ bias2,
    float* __restrict__ G) {
  const int dir = blockIdx.z;
  const _Float16* B = wih_h + (size_t)dir * GATES * KPAD;
  const float* bias = bias2 + dir * GATES;
  float* Gd = G + (size_t)dir * S_LEN * GATES;
  const int tid = threadIdx.x, lane = tid & 31, wave = tid >> 5;
  const int m0 = blockIdx.y * 128 + (wave & 3) * 32;   // 4 waves along M
  const int n0 = blockIdx.x * 64 + (wave >> 2) * 32;   // 2 waves along N
  v8f acc00 = {}, acc01 = {}, acc10 = {}, acc11 = {};
  for (int k0 = 0; k0 < KPAD; k0 += 32) {
    if (k0 + 32 < KPAD) {           // prefetch next K panel (global_prefetch_b8)
      __builtin_prefetch(A + (size_t)(m0 + (lane & 15)) * KPAD + k0 + 32, 0, 1);
      __builtin_prefetch(B + (size_t)(n0 + (lane & 15)) * KPAD + k0 + 32, 0, 1);
    }
    v16h a0 = load_a_frag(A, KPAD, m0,      k0, lane);
    v16h a1 = load_a_frag(A, KPAD, m0 + 16, k0, lane);
    v16h b0 = load_b_frag(B, KPAD, n0,      k0, lane);
    v16h b1 = load_b_frag(B, KPAD, n0 + 16, k0, lane);
    acc00 = __builtin_amdgcn_wmma_f32_16x16x32_f16(false, a0, false, b0, (short)0, acc00, false, false);
    acc01 = __builtin_amdgcn_wmma_f32_16x16x32_f16(false, a0, false, b1, (short)0, acc01, false, false);
    acc10 = __builtin_amdgcn_wmma_f32_16x16x32_f16(false, a1, false, b0, (short)0, acc10, false, false);
    acc11 = __builtin_amdgcn_wmma_f32_16x16x32_f16(false, a1, false, b1, (short)0, acc11, false, false);
  }
  store_tile(acc00, Gd, bias, m0,      n0,      lane);
  store_tile(acc01, Gd, bias, m0,      n0 + 16, lane);
  store_tile(acc10, Gd, bias, m0 + 16, n0,      lane);
  store_tile(acc11, Gd, bias, m0 + 16, n0 + 16, lane);
}

// ---------------------------------------------------------------------------
// Recurrent LSTM: 8 WGP-slices x 2 directions. Each slice owns 64 hidden
// units (256 gate rows), keeps its Whh slab in LDS (256 KB f16, layout
// [k-pair][row] -> bank-conflict-free), exchanges h through L2 with a
// double-buffered agent-scope barrier each step. Hidden states stored f16.
// ---------------------------------------------------------------------------
__device__ __forceinline__ float sigm(float x) { return 1.0f / (1.0f + __expf(-x)); }

__global__ __launch_bounds__(512) void lstm_kernel(const _Float16* __restrict__ whh_t,
    const float* __restrict__ G, _Float16* __restrict__ lstm_out_h,
    float* __restrict__ h_exch, unsigned* __restrict__ flags) {
  extern __shared__ char smem[];
  v2h*   w_s = (v2h*)smem;                                   // [256 pairs][256 rows]
  float* h_s = (float*)(smem + 256 * 256 * sizeof(v2h));     // [512]
  float* g_s = h_s + HID;                                    // [256]

  const int dir = blockIdx.y, slice = blockIdx.x, tid = threadIdx.x;
  const int j0 = slice * 64;
  const _Float16* W = whh_t + (size_t)dir * HID * GATES;     // [k][n]
  const float* Gd = G + (size_t)dir * S_LEN * GATES;
  float* hx = h_exch + (size_t)dir * 2 * HID;                // double buffer [2][512]
  unsigned* cnt = flags + dir * 16;                          // separate cachelines

  // stage Whh slice into LDS: rows r = gate*64 + (j - j0)
  for (int idx = tid; idx < 256 * 256; idx += 512) {
    int p = idx >> 8, r = idx & 255;
    int col = (r >> 6) * HID + j0 + (r & 63);
    v2h w;
    w.x = W[(size_t)(2 * p)     * GATES + col];
    w.y = W[(size_t)(2 * p + 1) * GATES + col];
    w_s[idx] = w;
  }
  h_s[tid] = 0.0f;                   // blockDim == HID
  float cj = 0.0f;                   // cell state, owned by tid < 64
  __syncthreads();

  const int r = tid >> 1, half = tid & 1, pbase = half * 128;
  for (int step = 0; step < S_LEN; ++step) {
    const int t = dir ? (S_LEN - 1 - step) : step;
    // 256-row matvec, 2 threads per row (k halves), f16 weights from LDS
    float s = 0.0f;
    #pragma unroll 4
    for (int p = 0; p < 128; ++p) {
      v2h w = w_s[(pbase + p) * 256 + r];
      int k = (pbase + p) * 2;
      s += (float)w.x * h_s[k] + (float)w.y * h_s[k + 1];
    }
    s += __shfl_xor(s, 1, 32);
    if (!half) g_s[r] = s;
    __syncthreads();

    if (tid < 64) {
      const int j = tid;
      const size_t gb = (size_t)t * GATES + j0 + j;
      float pi = g_s[j]        + Gd[gb];
      float pf = g_s[64 + j]   + Gd[gb + 512];
      float pg = g_s[128 + j]  + Gd[gb + 1024];
      float po = g_s[192 + j]  + Gd[gb + 1536];
      cj = sigm(pf) * cj + sigm(pi) * tanhf(pg);
      float h = sigm(po) * tanhf(cj);
      lstm_out_h[(size_t)t * 1024 + dir * HID + j0 + j] = (_Float16)h;
      __hip_atomic_store(&hx[(step & 1) * HID + j0 + j], h,
                         __ATOMIC_RELAXED, __HIP_MEMORY_SCOPE_AGENT);
      __threadfence();
    }
    __syncthreads();
    if (tid == 0) {                  // arrive + spin (agent scope)
      __hip_atomic_fetch_add(cnt, 1u, __ATOMIC_RELEASE, __HIP_MEMORY_SCOPE_AGENT);
      const unsigned target = (unsigned)NSLICE * (step + 1);
      while (__hip_atomic_load(cnt, __ATOMIC_ACQUIRE, __HIP_MEMORY_SCOPE_AGENT) < target)
        __builtin_amdgcn_s_sleep(1);
    }
    __syncthreads();
    h_s[tid] = __hip_atomic_load(&hx[(step & 1) * HID + tid],
                                 __ATOMIC_RELAXED, __HIP_MEMORY_SCOPE_AGENT);
    __syncthreads();
  }
}

// ---------------------------------------------------------------------------
// hidden2tag as WMMA GEMM: feats[4096][11] = lstm_out_h[4096][1024] @
// h2t_h^T[1024][16], N padded to 16, K=1024 -> 32 wmma steps per wave.
// ---------------------------------------------------------------------------
__global__ __launch_bounds__(256) void tag_gemm_kernel(const _Float16* __restrict__ A,
    const _Float16* __restrict__ h2t_h, const float* __restrict__ b,
    float* __restrict__ feats) {
  const int tid = threadIdx.x, lane = tid & 31, wave = tid >> 5;
  const int m0 = blockIdx.x * 128 + wave * 16;      // 8 waves, 16 rows each
  v8f acc = {};
  for (int k0 = 0; k0 < 2 * HID; k0 += 32) {
    v16h a  = load_a_frag(A, 2 * HID, m0, k0, lane);
    v16h bf = load_b_frag(h2t_h, 2 * HID, 0, k0, lane);
    acc = __builtin_amdgcn_wmma_f32_16x16x32_f16(false, a, false, bf, (short)0, acc, false, false);
  }
  const int n = lane & 15;
  const int mbase = m0 + ((lane & 16) ? 8 : 0);
  if (n < TAGS) {
    float bv = b[n];
    #pragma unroll
    for (int r = 0; r < 8; ++r) feats[(size_t)(mbase + r) * TAGS + n] = acc[r] + bv;
  }
}

// ---------------------------------------------------------------------------
// Viterbi decode (sequential, single block). out[0]=score, out[1+t]=path[t].
// ---------------------------------------------------------------------------
__global__ __launch_bounds__(128) void viterbi_kernel(const float* __restrict__ feats,
    const float* __restrict__ transitions, int* __restrict__ bptr, float* __restrict__ out) {
  __shared__ float tr[TAGS][TAGS];
  __shared__ float fv[TAGS];
  const int tid = threadIdx.x;
  if (tid < TAGS * TAGS) tr[tid / TAGS][tid % TAGS] = transitions[tid];
  if (tid < TAGS) fv[tid] = (tid == START_TAG) ? 0.0f : NEGV;
  __syncthreads();
  for (int t = 0; t < S_LEN; ++t) {
    float best = -3.0e38f; int bj = 0;
    if (tid < TAGS) {
      #pragma unroll
      for (int j = 0; j < TAGS; ++j) {
        float v = fv[j] + tr[tid][j];
        if (v > best) { best = v; bj = j; }      // first-max, matches argmax
      }
      bptr[t * TAGS + tid] = bj;
      best += feats[t * TAGS + tid];
    }
    __syncthreads();
    if (tid < TAGS) fv[tid] = best;
    __syncthreads();
  }
  if (tid == 0) {
    float bestv = -3.0e38f; int bi = 0;
    for (int i = 0; i < TAGS; ++i) {
      float v = fv[i] + tr[STOP_TAG][i];
      if (i == START_TAG || i == STOP_TAG) v = NEGV;
      if (v > bestv) { bestv = v; bi = i; }
    }
    out[0] = bestv;
    int tag = bi;
    for (int t = S_LEN - 1; t >= 0; --t) {
      out[1 + t] = (float)tag;
      tag = bptr[t * TAGS + tag];
    }
  }
}

// ---------------------------------------------------------------------------
extern "C" void kernel_launch(void* const* d_in, const int* in_sizes, int n_in,
                              void* d_out, int out_size, void* d_ws, size_t ws_size,
                              hipStream_t stream) {
  (void)in_sizes; (void)n_in; (void)out_size;
  const int*   sentence = (const int*)d_in[0];
  const int*   chars    = (const int*)d_in[1];
  const float* word_emb = (const float*)d_in[4];
  const float* char_emb = (const float*)d_in[5];
  const float* conv_w   = (const float*)d_in[6];
  const float* conv_b   = (const float*)d_in[7];
  const float* Wih_f = (const float*)d_in[8];
  const float* Whh_f = (const float*)d_in[9];
  const float* bih_f = (const float*)d_in[10];
  const float* bhh_f = (const float*)d_in[11];
  const float* Wih_b = (const float*)d_in[12];
  const float* Whh_b = (const float*)d_in[13];
  const float* bih_b = (const float*)d_in[14];
  const float* bhh_b = (const float*)d_in[15];
  const float* h2t_w = (const float*)d_in[16];
  const float* h2t_b = (const float*)d_in[17];
  const float* transitions = (const float*)d_in[18];
  float* out = (float*)d_out;

  char* ws = (char*)d_ws;
  size_t off = 0;
  auto wsAlloc = [&](size_t bytes) -> void* {
    void* p = ws + off; off += (bytes + 255) & ~(size_t)255; return p;
  };
  _Float16* emb_h    = (_Float16*)wsAlloc((size_t)S_LEN * KPAD * 2);
  _Float16* wih_h    = (_Float16*)wsAlloc(2ULL * GATES * KPAD * 2);
  _Float16* whh_t    = (_Float16*)wsAlloc(2ULL * HID * GATES * 2);
  float*    bias2    = (float*)wsAlloc(2ULL * GATES * 4);
  float*    G        = (float*)wsAlloc(2ULL * S_LEN * GATES * 4);
  _Float16* lstmout  = (_Float16*)wsAlloc((size_t)S_LEN * 1024 * 2);
  _Float16* h2t_h    = (_Float16*)wsAlloc((size_t)NPAD * 2 * HID * 2);
  float*    feats    = (float*)wsAlloc((size_t)S_LEN * TAGS * 4);
  int*      bptr     = (int*)wsAlloc((size_t)S_LEN * TAGS * 4);
  float*    h_exch   = (float*)wsAlloc(2ULL * 2 * HID * 4);
  unsigned* flags    = (unsigned*)wsAlloc(256);
  if (off > ws_size) return;

  hipMemsetAsync(flags, 0, 256, stream);   // reset barrier counters every call

  convert_kernel<<<1024, 256, 0, stream>>>(Wih_f, Whh_f, bih_f, bhh_f,
                                           Wih_b, Whh_b, bih_b, bhh_b, h2t_w,
                                           wih_h, whh_t, bias2, h2t_h);
  prep_kernel<<<S_LEN, 128, 0, stream>>>(sentence, chars, word_emb, char_emb,
                                         conv_w, conv_b, emb_h);
  gemm_gates_kernel<<<dim3(GATES / 64, S_LEN / 128, 2), 256, 0, stream>>>(
      emb_h, wih_h, bias2, G);

  const size_t smem = 256 * 256 * sizeof(v2h) + HID * 4 + 256 * 4;  // ~265 KB
  (void)hipFuncSetAttribute((const void*)lstm_kernel,
                            hipFuncAttributeMaxDynamicSharedMemorySize, (int)smem);
  lstm_kernel<<<dim3(NSLICE, 2), HID, smem, stream>>>(whh_t, G, lstmout, h_exch, flags);

  tag_gemm_kernel<<<S_LEN / 128, 256, 0, stream>>>(lstmout, h2t_h, h2t_b, feats);
  viterbi_kernel<<<1, 128, 0, stream>>>(feats, transitions, bptr, out);
}
